// attention_block_9887014715421
// MI455X (gfx1250) — matrix operand order repaired and checked by reading, not compiled, not run on hardware
//
#include <hip/hip_runtime.h>
#include <hip/hip_bf16.h>
#include <math.h>

// ---------------------------------------------------------------------------
// attention-over-width block: 1x1 conv QKV proj -> scaled QK^T (+causal mask,
// scores emitted) -> softmax -> AV.  All matmuls via v_wmma_f32_16x16x32_f16.
//   B=8, C_in=4, C_out=64, H=8, W=2048, d = C_out*H = 512
// Staging (f16, in d_ws):
//   Qt [b][pos][d]  (pos-major, A-fragments for QK^T)
//   Kt [b][pos][d]  (pos-major, B-fragments for QK^T)
//   Vd [b][d][pos]  (d-major,  A-fragments for AV)
//   At [b][k][l]    (attn,     B-fragments for AV)
// ---------------------------------------------------------------------------

typedef __attribute__((ext_vector_type(16))) _Float16 v16h;
typedef __attribute__((ext_vector_type(8)))  _Float16 v8h;
typedef __attribute__((ext_vector_type(8)))  float    v8f;

#define BATCH 8
#define CIN   4
#define COUT  64
#define HGT   8
#define WID   2048
#define DDIM  512   // COUT*HGT

// ---------------- stage 1: 1x1 conv projections, f32 -> f16 ----------------
__global__ void __launch_bounds__(256)
proj_qkv(const float* __restrict__ x,
         const float* __restrict__ Wq, const float* __restrict__ bq,
         const float* __restrict__ Wk, const float* __restrict__ bk,
         const float* __restrict__ Wv, const float* __restrict__ bv,
         _Float16* __restrict__ Qt, _Float16* __restrict__ Kt,
         _Float16* __restrict__ Vd) {
  int tid = blockIdx.x * blockDim.x + threadIdx.x;
  if (tid >= BATCH * COUT * WID) return;
  int l = tid % WID;
  int o = (tid / WID) % COUT;
  int b = tid / (WID * COUT);

  float wqr[CIN], wkr[CIN], wvr[CIN];
#pragma unroll
  for (int c = 0; c < CIN; ++c) {
    wqr[c] = Wq[o * CIN + c];
    wkr[c] = Wk[o * CIN + c];
    wvr[c] = Wv[o * CIN + c];
  }
  const float bqv = bq[o], bkv = bk[o], bvv = bv[o];

  v8h qv, kv;
#pragma unroll
  for (int hh = 0; hh < HGT; ++hh) {
    float q = bqv, k = bkv, v = bvv;
#pragma unroll
    for (int c = 0; c < CIN; ++c) {
      float xv = x[(((size_t)b * CIN + c) * HGT + hh) * WID + l];
      q += xv * wqr[c];
      k += xv * wkr[c];
      v += xv * wvr[c];
    }
    qv[hh] = (_Float16)q;
    kv[hh] = (_Float16)k;
    Vd[((size_t)b * DDIM + o * HGT + hh) * WID + l] = (_Float16)v;
  }
  size_t base = ((size_t)b * WID + l) * DDIM + (size_t)o * HGT; // 16B aligned
  *(v8h*)(Qt + base) = qv;
  *(v8h*)(Kt + base) = kv;
}

// ------------- stage 2: scores = (Q^T K)/sqrt(d), causal mask --------------
// one wave per 16x16 score tile; 8 waves / block.
__global__ void __launch_bounds__(256)
scores_qk(const _Float16* __restrict__ Qt, const _Float16* __restrict__ Kt,
          float* __restrict__ scores, const int* __restrict__ causalp) {
  const int b    = blockIdx.y;
  const int wave = threadIdx.x >> 5;
  const int lane = threadIdx.x & 31;
  const int tileIdx = blockIdx.x * 8 + wave;  // 0 .. 128*128-1
  const int k0 = (tileIdx >> 7) * 16;
  const int l0 = (tileIdx & 127) * 16;
  const int hh  = lane >> 4;   // half-select
  const int sub = lane & 15;   // A: m   B: n
  const int causal = *causalp;

  const _Float16* qrow = Qt + ((size_t)b * WID + (k0 + sub)) * DDIM;
  const _Float16* krow = Kt + ((size_t)b * WID + (l0 + sub)) * DDIM;

  v8f acc = {};
  for (int d0 = 0; d0 < DDIM; d0 += 32) {
    v16h a = *(const v16h*)(qrow + d0 + 16 * hh);
    v16h bf = *(const v16h*)(krow + d0 + 16 * hh);
    acc = __builtin_amdgcn_wmma_f32_16x16x32_f16(
        false, a, false, bf, (short)0, acc, false, false);
  }

  const float scale = 0.04419417382415922f;  // 1/sqrt(512)
  const int l = l0 + sub;
#pragma unroll
  for (int r = 0; r < 8; ++r) {
    int k = k0 + 8 * hh + r;
    float s = acc[r] * scale;
    if (causal) {
      s = (l >= k) ? s : 0.0f;              // jnp.triu
      s = (s != 0.0f) ? s : -INFINITY;      // where(!=0, s, -inf)
    }
    scores[((size_t)b * WID + k) * WID + l] = s;
  }
}

// ---------------- stage 3: row softmax, write attn as f16 ------------------
__global__ void __launch_bounds__(256)
softmax_rows(const float* __restrict__ scores, _Float16* __restrict__ attn) {
  const int row = blockIdx.x;  // 0 .. BATCH*WID-1
  const float* srow = scores + (size_t)row * WID;
  _Float16* arow = attn + (size_t)row * WID;
  __shared__ float red[256];

  float vals[8];
  float m = -INFINITY;
#pragma unroll
  for (int i = 0; i < 8; ++i) {
    vals[i] = srow[threadIdx.x + i * 256];
    m = fmaxf(m, vals[i]);
  }
  red[threadIdx.x] = m;
  __syncthreads();
  for (int s = 128; s > 0; s >>= 1) {
    if (threadIdx.x < s)
      red[threadIdx.x] = fmaxf(red[threadIdx.x], red[threadIdx.x + s]);
    __syncthreads();
  }
  m = red[0];
  __syncthreads();

  float sum = 0.f;
#pragma unroll
  for (int i = 0; i < 8; ++i) { vals[i] = __expf(vals[i] - m); sum += vals[i]; }
  red[threadIdx.x] = sum;
  __syncthreads();
  for (int s = 128; s > 0; s >>= 1) {
    if (threadIdx.x < s) red[threadIdx.x] += red[threadIdx.x + s];
    __syncthreads();
  }
  const float inv = 1.0f / red[0];
#pragma unroll
  for (int i = 0; i < 8; ++i)
    arow[threadIdx.x + i * 256] = (_Float16)(vals[i] * inv);
}

// ---------------- stage 4: x_out[d,k] = sum_l V[d,l] * attn[k,l] -----------
// one wave per 16(d) x 16(k) output tile; causal lets us skip l-chunks < k0.
__global__ void __launch_bounds__(256)
av_out(const _Float16* __restrict__ Vd, const _Float16* __restrict__ attn,
       float* __restrict__ xout, const int* __restrict__ causalp) {
  const int b    = blockIdx.y;
  const int wave = threadIdx.x >> 5;
  const int lane = threadIdx.x & 31;
  const int tileIdx = blockIdx.x * 8 + wave;  // 0 .. 32*128-1
  const int d0 = (tileIdx >> 7) * 16;
  const int k0 = (tileIdx & 127) * 16;
  const int hh  = lane >> 4;
  const int sub = lane & 15;
  const int causal = *causalp;

  const _Float16* vrow = Vd   + ((size_t)b * DDIM + (d0 + sub)) * WID; // A: m
  const _Float16* arow = attn + ((size_t)b * WID  + (k0 + sub)) * WID; // B: n

  v8f acc = {};
  const int lstart = causal ? ((k0 >> 5) << 5) : 0;  // chunks fully masked out
  for (int l0 = lstart; l0 < WID; l0 += 32) {
    v16h a = *(const v16h*)(vrow + l0 + 16 * hh);
    v16h bf = *(const v16h*)(arow + l0 + 16 * hh);
    acc = __builtin_amdgcn_wmma_f32_16x16x32_f16(
        false, a, false, bf, (short)0, acc, false, false);
  }

  const int k = k0 + sub;
#pragma unroll
  for (int r = 0; r < 8; ++r) {
    int d = d0 + 8 * hh + r;
    xout[((size_t)b * DDIM + d) * WID + k] = acc[r];
  }
}

// ---------------------------------------------------------------------------
extern "C" void kernel_launch(void* const* d_in, const int* in_sizes, int n_in,
                              void* d_out, int out_size, void* d_ws, size_t ws_size,
                              hipStream_t stream) {
  const float* x  = (const float*)d_in[0];
  const float* Wq = (const float*)d_in[1];
  const float* bq = (const float*)d_in[2];
  const float* Wk = (const float*)d_in[3];
  const float* bk = (const float*)d_in[4];
  const float* Wv = (const float*)d_in[5];
  const float* bv = (const float*)d_in[6];
  const int* causal = (const int*)d_in[7];

  float* xout   = (float*)d_out;                                  // [8,64,8,2048]
  float* scores = xout + (size_t)BATCH * DDIM * WID;              // [8,2048,2048]

  const size_t nQ = (size_t)BATCH * WID * DDIM;   // 8,388,608 halfs each
  const size_t nA = (size_t)BATCH * WID * WID;    // 33,554,432 halfs
  if (ws_size < (3 * nQ + nA) * sizeof(_Float16)) return;  // ~112 MB needed

  _Float16* Qt = (_Float16*)d_ws;
  _Float16* Kt = Qt + nQ;
  _Float16* Vd = Kt + nQ;
  _Float16* At = Vd + nQ;

  const int nproj = BATCH * COUT * WID;
  proj_qkv<<<(nproj + 255) / 256, 256, 0, stream>>>(x, Wq, bq, Wk, bk, Wv, bv,
                                                    Qt, Kt, Vd);
  // 128*128 tiles per batch, 8 waves (tiles) per 256-thread block
  scores_qk<<<dim3((128 * 128) / 8, BATCH), 256, 0, stream>>>(Qt, Kt, scores,
                                                              causal);
  softmax_rows<<<BATCH * WID, 256, 0, stream>>>(scores, At);
  // 32*128 tiles per batch
  av_out<<<dim3((32 * 128) / 8, BATCH), 256, 0, stream>>>(Vd, At, xout, causal);
}
